// MeinBlock_41781441855658
// MI455X (gfx1250) — matrix operand order repaired and checked by reading, not compiled, not run on hardware
//
#include <hip/hip_runtime.h>
#include <hip/hip_bf16.h>
#include <limits.h>

// ---------------------------------------------------------------------------
// GATv2 x2 + BN + PReLU + residual for MI455X (gfx1250), wave32, WMMA f16.
// ---------------------------------------------------------------------------

#define D_CH 256
#define SLOPE 0.2f
#define BN_EPS 1e-5f

typedef _Float16 v16h __attribute__((ext_vector_type(16)));
typedef _Float16 v8h  __attribute__((ext_vector_type(8)));
typedef float    v8f  __attribute__((ext_vector_type(8)));

union V16 { v16h v; v8h h[2]; };

// ---------------- utility kernels ----------------

__global__ void k_zero_f(float* __restrict__ p, long n) {
    long i = (long)blockIdx.x * blockDim.x + threadIdx.x;
    if (i < n) p[i] = 0.0f;
}

__global__ void k_fill_i(int* __restrict__ p, int v, long n) {
    long i = (long)blockIdx.x * blockDim.x + threadIdx.x;
    if (i < n) p[i] = v;
}

__global__ void k_f2h(const float* __restrict__ x, _Float16* __restrict__ xh, long n) {
    long i = (long)blockIdx.x * blockDim.x + threadIdx.x;
    if (i < n) xh[i] = (_Float16)x[i];
}

// WT[n*256+k] = (f16) W[k*256+n]   (transpose so a B-fragment column is contiguous)
__global__ void k_wtrans(const float* __restrict__ W, _Float16* __restrict__ WT) {
    int k = blockIdx.x * 16 + threadIdx.x;
    int n = blockIdx.y * 16 + threadIdx.y;
    WT[(size_t)n * D_CH + k] = (_Float16)W[(size_t)k * D_CH + n];
}

// ---------------- self-loop mean edge-attr ----------------

__global__ void k_deg(const int* __restrict__ dst, const float* __restrict__ eattr,
                      float* __restrict__ deg, float* __restrict__ suma, int E) {
    int e = blockIdx.x * blockDim.x + threadIdx.x;
    if (e >= E) return;
    int d = dst[e];
    atomicAdd(&deg[d], 1.0f);
    atomicAdd(&suma[2 * d + 0], eattr[2 * e + 0]);
    atomicAdd(&suma[2 * d + 1], eattr[2 * e + 1]);
}

__global__ void k_mean(const float* __restrict__ deg, float* __restrict__ suma, int N) {
    int n = blockIdx.x * blockDim.x + threadIdx.x;
    if (n >= N) return;
    float dg = fmaxf(deg[n], 1.0f);
    suma[2 * n + 0] /= dg;
    suma[2 * n + 1] /= dg;
}

// ---------------- dual GEMM: C_a = A*Wa, C_b = A*Wb (f16 in, f32 out) -------
// Each wave: 16 rows x 64 cols for BOTH weights (shared A fragment).
// Block = 128 threads (4 waves) -> covers all 256 columns of one 16-row strip.
// waves_per_eu(1) relaxes the register budget so the scheduler can keep many
// B fragments in flight; the K loop is fully unrolled (constant offsets) so
// loads can be hoisted/pipelined across K-steps.

__global__ __launch_bounds__(128)
__attribute__((amdgpu_waves_per_eu(1)))
void k_gemm_dual(const _Float16* __restrict__ Ah,
                 const _Float16* __restrict__ WTa,   // [256 cols][256 K] f16 (transposed)
                 const _Float16* __restrict__ WTb,
                 float* __restrict__ Ca, float* __restrict__ Cb, int nrows) {
    const int wave = threadIdx.x >> 5;
    const int lane = threadIdx.x & 31;
    const int row0 = blockIdx.x * 16;
    if (row0 >= nrows) return;

    const int M     = lane & 15;                 // A row / B column within tile
    const int hiHlf = lane >> 4;                 // 0: lanes 0-15, 1: lanes 16-31
    const int koffA = hiHlf * 16;                // A: 16 contiguous halfs at kb+{0|16}
    const int koffB = hiHlf * 16;                // B: 16 contiguous halfs at kb+{0|16}
    const int n0    = wave * 64;

    v8f accA[4] = {};
    v8f accB[4] = {};

    const _Float16* paBase = Ah + (size_t)(row0 + M) * D_CH + koffA;
    // per-column-tile base pointers (constant offsets below after unroll)
    const _Float16* pba[4];
    const _Float16* pbb[4];
#pragma unroll
    for (int t = 0; t < 4; ++t) {
        const size_t cofs = (size_t)(n0 + t * 16 + M) * D_CH + koffB;
        pba[t] = WTa + cofs;
        pbb[t] = WTb + cofs;
    }

#pragma unroll
    for (int kt = 0; kt < 8; ++kt) {
        const int kb = kt * 32;

        V16 a;
        a.h[0] = *(const v8h*)(paBase + kb);
        a.h[1] = *(const v8h*)(paBase + kb + 8);

        // all 8 B fragments for this K-step in independent registers
        V16 ba[4], bb[4];
#pragma unroll
        for (int t = 0; t < 4; ++t) {
            ba[t].h[0] = *(const v8h*)(pba[t] + kb);
            ba[t].h[1] = *(const v8h*)(pba[t] + kb + 8);
            bb[t].h[0] = *(const v8h*)(pbb[t] + kb);
            bb[t].h[1] = *(const v8h*)(pbb[t] + kb + 8);
        }

#pragma unroll
        for (int t = 0; t < 4; ++t) {
            accA[t] = __builtin_amdgcn_wmma_f32_16x16x32_f16(
                false, a.v, false, ba[t].v, (short)0, accA[t], false, false);
            accB[t] = __builtin_amdgcn_wmma_f32_16x16x32_f16(
                false, a.v, false, bb[t].v, (short)0, accB[t], false, false);
        }
    }

    // D layout: element r -> row (r + hiHlf*8), col lane&15
    const int rowHi = hiHlf * 8;
    const int col   = lane & 15;
#pragma unroll
    for (int t = 0; t < 4; ++t) {
#pragma unroll
        for (int r = 0; r < 8; ++r) {
            int row = row0 + r + rowHi;
            if (row < nrows) {
                size_t idx = (size_t)row * D_CH + n0 + t * 16 + col;
                Ca[idx] = accA[t][r];
                Cb[idx] = accB[t][r];
            }
        }
    }
}

// ---------------- edge kernels ----------------

__device__ __forceinline__ int   fenc(float f) { int i = __float_as_int(f); return (i >= 0) ? i : (i ^ 0x7fffffff); }
__device__ __forceinline__ float fdec(int i)   { return __int_as_float((i >= 0) ? i : (i ^ 0x7fffffff)); }

// one wave per edge: logit = att . leaky_relu(xl[src] + xr[dst] + eattr@We)
__global__ __launch_bounds__(256)
void k_edge_logits(const int* __restrict__ src, const int* __restrict__ dst,
                   const float* __restrict__ eattr, const float* __restrict__ meanattr,
                   const float* __restrict__ xl, const float* __restrict__ xr,
                   const float* __restrict__ We, const float* __restrict__ att,
                   float* __restrict__ logits, int* __restrict__ mkey, int E, int N) {
    long e = (long)blockIdx.x * (blockDim.x >> 5) + (threadIdx.x >> 5);
    if (e >= (long)E + N) return;
    int lane = threadIdx.x & 31;
    int s, d; float ea0, ea1;
    if (e < E) {
        s = src[e]; d = dst[e];
        ea0 = eattr[2 * e + 0]; ea1 = eattr[2 * e + 1];
    } else {
        s = d = (int)(e - E);
        ea0 = meanattr[2 * s + 0]; ea1 = meanattr[2 * s + 1];
    }
    const float* pl = xl + (size_t)s * D_CH;
    const float* pr = xr + (size_t)d * D_CH;
    float acc = 0.0f;
#pragma unroll
    for (int j = 0; j < 8; ++j) {
        int c = lane + 32 * j;
        float v = pl[c] + pr[c] + ea0 * We[c] + ea1 * We[D_CH + c];
        v = (v > 0.0f) ? v : SLOPE * v;
        acc += v * att[c];
    }
#pragma unroll
    for (int off = 16; off > 0; off >>= 1) acc += __shfl_xor(acc, off, 32);
    if (lane == 0) {
        logits[e] = acc;
        atomicMax(&mkey[d], fenc(acc));
    }
}

// one thread per edge: ex = exp(logit - max[dst]); denom[dst] += ex
__global__ void k_edge_exp(const int* __restrict__ dst, const float* __restrict__ logits,
                           const int* __restrict__ mkey, float* __restrict__ ex,
                           float* __restrict__ denom, int E, int N) {
    long e = (long)blockIdx.x * blockDim.x + threadIdx.x;
    if (e >= (long)E + N) return;
    int d = (e < E) ? dst[e] : (int)(e - E);
    float v = __expf(logits[e] - fdec(mkey[d]));
    ex[e] = v;
    atomicAdd(&denom[d], v);
}

// one wave per edge: acc[dst] += xl[src] * alpha
__global__ __launch_bounds__(256)
void k_edge_scatter(const int* __restrict__ src, const int* __restrict__ dst,
                    const float* __restrict__ xl, const float* __restrict__ ex,
                    const float* __restrict__ denom, float* __restrict__ acc,
                    int E, int N) {
    long e = (long)blockIdx.x * (blockDim.x >> 5) + (threadIdx.x >> 5);
    if (e >= (long)E + N) return;
    int lane = threadIdx.x & 31;
    int s, d;
    if (e < E) { s = src[e]; d = dst[e]; }
    else       { s = d = (int)(e - E); }
    float alpha = ex[e] / denom[d];
    const float* pl = xl + (size_t)s * D_CH;
    float* po = acc + (size_t)d * D_CH;
#pragma unroll
    for (int j = 0; j < 8; ++j) {
        int c = lane + 32 * j;
        atomicAdd(&po[c], pl[c] * alpha);
    }
}

// ---------------- batchnorm ----------------

// each block: 64 rows, thread t = channel t; coalesced column reduction
__global__ __launch_bounds__(256)
void k_bn_stats(const float* __restrict__ acc, const float* __restrict__ bias,
                float* __restrict__ sums, float* __restrict__ sumsq, int nrows) {
    int t = threadIdx.x;
    int row0 = blockIdx.x * 64;
    float b = bias[t];
    float s = 0.0f, s2 = 0.0f;
    for (int r = 0; r < 64; ++r) {
        int row = row0 + r;
        if (row < nrows) {
            float v = acc[(size_t)row * D_CH + t] + b;
            s += v; s2 += v * v;
        }
    }
    atomicAdd(&sums[t], s);
    atomicAdd(&sumsq[t], s2);
}

__global__ void k_bn_final(const float* __restrict__ sums, const float* __restrict__ sumsq,
                           const float* __restrict__ g, const float* __restrict__ beta,
                           float* __restrict__ scale, float* __restrict__ shift, float invN) {
    int c = threadIdx.x;
    float mu = sums[c] * invN;
    float var = sumsq[c] * invN - mu * mu;
    float sc = g[c] * rsqrtf(var + BN_EPS);
    scale[c] = sc;
    shift[c] = beta[c] - mu * sc;
}

// layer-1 epilogue: h = prelu(bn(acc + b)), emit as f16 for layer-2 GEMM
__global__ void k_bn_apply_h(const float* __restrict__ acc, const float* __restrict__ bias,
                             const float* __restrict__ scale, const float* __restrict__ shift,
                             const float* __restrict__ aP, _Float16* __restrict__ outh, long n) {
    long i = (long)blockIdx.x * blockDim.x + threadIdx.x;
    if (i >= n) return;
    int c = (int)(i & (D_CH - 1));
    float v = (acc[i] + bias[c]) * scale[c] + shift[c];
    float a = aP[0];
    v = (v >= 0.0f) ? v : a * v;
    outh[i] = (_Float16)v;
}

// layer-2 epilogue: out = prelu(x + bn(acc + b))
__global__ void k_final(const float* __restrict__ acc, const float* __restrict__ bias,
                        const float* __restrict__ scale, const float* __restrict__ shift,
                        const float* __restrict__ aP, const float* __restrict__ xres,
                        float* __restrict__ out, long n) {
    long i = (long)blockIdx.x * blockDim.x + threadIdx.x;
    if (i >= n) return;
    int c = (int)(i & (D_CH - 1));
    float v = (acc[i] + bias[c]) * scale[c] + shift[c];
    float r = xres[i] + v;
    float a = aP[0];
    out[i] = (r >= 0.0f) ? r : a * r;
}

// ---------------------------------------------------------------------------

extern "C" void kernel_launch(void* const* d_in, const int* in_sizes, int n_in,
                              void* d_out, int out_size, void* d_ws, size_t ws_size,
                              hipStream_t stream) {
    const int N = in_sizes[0] / D_CH;     // 50000
    const int E = in_sizes[1] / 2;        // 800000
    const long ND = (long)N * D_CH;
    const long EN = (long)E + N;

    const float* x      = (const float*)d_in[0];
    const int*   src    = (const int*)d_in[1];
    const int*   dst    = src + E;
    const float* eattr  = (const float*)d_in[2];
    const float* Wl1 = (const float*)d_in[3],  *Wr1 = (const float*)d_in[4];
    const float* We1 = (const float*)d_in[5],  *att1 = (const float*)d_in[6];
    const float* b1  = (const float*)d_in[7];
    const float* Wl2 = (const float*)d_in[8],  *Wr2 = (const float*)d_in[9];
    const float* We2 = (const float*)d_in[10], *att2 = (const float*)d_in[11];
    const float* b2  = (const float*)d_in[12];
    const float* g1 = (const float*)d_in[13], *beta1 = (const float*)d_in[14];
    const float* g2 = (const float*)d_in[15], *beta2 = (const float*)d_in[16];
    const float* a1 = (const float*)d_in[17], *a2 = (const float*)d_in[18];

    // ---- workspace carve-up ----
    char* ws = (char*)d_ws;
    size_t off = 0;
    auto take = [&](size_t bytes) -> char* {
        char* p = ws + off;
        off = (off + bytes + 255) & ~(size_t)255;
        return p;
    };
    // contiguous zero-block: acc | denom | sums | sumsq
    float* accB  = (float*)take((ND + N + 512) * sizeof(float));
    float* denom = accB + ND;
    float* sums  = denom + N;
    float* sumsq = sums + D_CH;
    const long Z0 = ND + N + 512;
    // contiguous zero-block: deg | meanattr
    float* deg      = (float*)take((size_t)3 * N * sizeof(float));
    float* meanattr = deg + N;
    float* xl = (float*)take(ND * sizeof(float));
    float* xr = (float*)take(ND * sizeof(float));
    _Float16* xh  = (_Float16*)take(ND * sizeof(_Float16));
    _Float16* WT1l = (_Float16*)take((size_t)D_CH * D_CH * 2);
    _Float16* WT1r = (_Float16*)take((size_t)D_CH * D_CH * 2);
    _Float16* WT2l = (_Float16*)take((size_t)D_CH * D_CH * 2);
    _Float16* WT2r = (_Float16*)take((size_t)D_CH * D_CH * 2);
    int*   mkey   = (int*)take((size_t)N * sizeof(int));
    float* logits = (float*)take((size_t)EN * sizeof(float));
    float* ex     = (float*)take((size_t)EN * sizeof(float));
    float* scale  = (float*)take(D_CH * sizeof(float));
    float* shift  = (float*)take(D_CH * sizeof(float));
    (void)ws_size; (void)n_in; (void)out_size;

    const int B = 256;
    dim3 wgrid(16, 16), wblk(16, 16);
    const int edgeWaveBlocks = (int)((EN + 7) / 8);     // 8 waves / 256-thread block
    const int gemmBlocks = (N + 15) / 16;

    // ---- prep: self-loop mean attr, f16 input, transposed f16 weights ----
    k_zero_f<<<(3 * (long)N + B - 1) / B, B, 0, stream>>>(deg, 3 * (long)N);
    k_deg<<<(E + B - 1) / B, B, 0, stream>>>(dst, eattr, deg, meanattr, E);
    k_mean<<<(N + B - 1) / B, B, 0, stream>>>(deg, meanattr, N);
    k_f2h<<<(ND + B - 1) / B, B, 0, stream>>>(x, xh, ND);
    k_wtrans<<<wgrid, wblk, 0, stream>>>(Wl1, WT1l);
    k_wtrans<<<wgrid, wblk, 0, stream>>>(Wr1, WT1r);
    k_wtrans<<<wgrid, wblk, 0, stream>>>(Wl2, WT2l);
    k_wtrans<<<wgrid, wblk, 0, stream>>>(Wr2, WT2r);

    for (int layer = 0; layer < 2; ++layer) {
        const _Float16* WTl = layer ? WT2l : WT1l;
        const _Float16* WTr = layer ? WT2r : WT1r;
        const float* We  = layer ? We2  : We1;
        const float* att = layer ? att2 : att1;
        const float* bb  = layer ? b2 : b1;
        const float* gg  = layer ? g2 : g1;
        const float* bt  = layer ? beta2 : beta1;

        k_zero_f<<<(Z0 + B - 1) / B, B, 0, stream>>>(accB, Z0);
        k_fill_i<<<(N + B - 1) / B, B, 0, stream>>>(mkey, INT_MIN, N);

        // xl = h @ Wl, xr = h @ Wr   (WMMA)
        k_gemm_dual<<<gemmBlocks, 128, 0, stream>>>(xh, WTl, WTr, xl, xr, N);

        k_edge_logits<<<edgeWaveBlocks, B, 0, stream>>>(src, dst, eattr, meanattr,
                                                        xl, xr, We, att, logits, mkey, E, N);
        k_edge_exp<<<(int)((EN + B - 1) / B), B, 0, stream>>>(dst, logits, mkey, ex, denom, E, N);
        k_edge_scatter<<<edgeWaveBlocks, B, 0, stream>>>(src, dst, xl, ex, denom, accB, E, N);

        k_bn_stats<<<(N + 63) / 64, B, 0, stream>>>(accB, bb, sums, sumsq, N);
        k_bn_final<<<1, D_CH, 0, stream>>>(sums, sumsq, gg, bt, scale, shift, 1.0f / (float)N);

        if (layer == 0) {
            // h1 -> f16 (reuses xh as layer-2 GEMM input)
            k_bn_apply_h<<<(ND + B - 1) / B, B, 0, stream>>>(accB, bb, scale, shift, a1, xh, ND);
        } else {
            k_final<<<(ND + B - 1) / B, B, 0, stream>>>(accB, bb, scale, shift, a2, x,
                                                        (float*)d_out, ND);
        }
    }
}
